// Seq2Seq_55989193670983
// MI455X (gfx1250) — matrix operand order repaired and checked by reading, not compiled, not run on hardware
//
#include <hip/hip_runtime.h>
#include <math.h>

#define B_   32
#define S_   128
#define T_   128
#define E_   128
#define U_   256
#define V_   32000
#define G4U  1024   // 4*U

typedef __bf16 bf16_t;
typedef bf16_t bf16x16 __attribute__((ext_vector_type(16)));
typedef float  f32x8   __attribute__((ext_vector_type(8)));
typedef unsigned int u32;
typedef u32 u32x8 __attribute__((ext_vector_type(8)));
typedef u32 u32x4 __attribute__((ext_vector_type(4)));

union AFrag { bf16x16 v; u32x8 u; };

// ---- WMMA fragment loaders (CDNA5 wave32 layouts, 16x16x32 bf16) ----------
// A (16x32, row-major, lda even): lane m=lane&15, g=lane>>4 holds K pairs
//   VGPR j: K = 2j + 16g  -> one contiguous 32B run per lane
__device__ __forceinline__ bf16x16 load_a_frag(const bf16_t* A, int lda) {
  const int lane = threadIdx.x & 31;
  const int m = lane & 15, g = lane >> 4;
  AFrag f;
#pragma unroll
  for (int j = 0; j < 8; ++j) {
    const int k = 2 * j + 16 * g;
    f.u[j] = *(const u32*)(A + (size_t)m * lda + k);
  }
  return f.v;
}

// B fragment from frag-major swizzled weights: block of 512 bf16 per
// (K-chunk of 32, N-tile of 16): [lane][16 bf16], lane holds column n=lane&15,
// rows k = 2j + 16*(lane>>4) (+1). One 32B contiguous load per lane.
__device__ __forceinline__ bf16x16 load_b_frag_swz(const bf16_t* base) {
  const int lane = threadIdx.x & 31;
  AFrag f;
  f.u = *(const u32x8*)(base + lane * 16);
  return f.v;
}

__device__ __forceinline__ f32x8 wmma_bf16(bf16x16 a, bf16x16 b, f32x8 c) {
  return __builtin_amdgcn_wmma_f32_16x16x32_bf16(false, a, false, b, (short)0, c,
                                                 false, false);
}

__device__ __forceinline__ float sigmf(float x) { return 1.0f / (1.0f + __expf(-x)); }

// ---- weight convert + swizzle: fp32 row-major [K][N] -> bf16 frag-major ----
// pair index p -> block = p>>8 = kc*(N/16)+ntile, lane = (p>>3)&31, j = p&7
__global__ void cvt_swz(const float* __restrict__ in, bf16_t* __restrict__ out,
                        int K, int N) {
  int p = blockIdx.x * 256 + threadIdx.x;     // one bf16 pair per thread
  int total = (K * N) >> 1;
  if (p >= total) return;
  int j = p & 7;
  int lane = (p >> 3) & 31;
  int blk = p >> 8;
  int nt = blk % (N >> 4);
  int kc = blk / (N >> 4);
  int g = lane >> 4, n16 = lane & 15;
  int k0 = kc * 32 + 2 * j + 16 * g;
  int n = nt * 16 + n16;
  out[(size_t)p * 2]     = (bf16_t)in[(size_t)k0 * N + n];
  out[(size_t)p * 2 + 1] = (bf16_t)in[(size_t)(k0 + 1) * N + n];
}

// ---- small helpers ---------------------------------------------------------
// enc_x[r = s*32+b][e] = enc_emb[src[b,s]][e]   (bf16)
__global__ void gather_enc(const int* __restrict__ src, const float* __restrict__ emb,
                           bf16_t* __restrict__ out) {
  int r = blockIdx.x;           // 0..4095
  int s = r >> 5, b = r & 31;
  int tok = src[b * S_ + s];
  out[(size_t)r * E_ + threadIdx.x] = (bf16_t)emb[(size_t)tok * E_ + threadIdx.x];
}

// dec tokens: t==0 -> tgt[b,0] else tgt[b,t-1]
__global__ void gather_dec(const int* __restrict__ tgt, const float* __restrict__ emb,
                           bf16_t* __restrict__ out) {
  int r = blockIdx.x;
  int t = r >> 5, b = r & 31;
  int tok = (t == 0) ? tgt[b * T_] : tgt[b * T_ + t - 1];
  out[(size_t)r * E_ + threadIdx.x] = (bf16_t)emb[(size_t)tok * E_ + threadIdx.x];
}

// keys [s*32+b][u] -> keysT [b][u][s]
__global__ void transpose_keys(const float* __restrict__ keys, float* __restrict__ keysT) {
  int r = blockIdx.x;
  int s = r >> 5, b = r & 31;
  int u = threadIdx.x;
  keysT[((size_t)b * U_ + u) * S_ + s] = keys[(size_t)r * U_ + u];
}

// ---- generic bf16 WMMA GEMM: C[M,N] = A[M,K]*Bswz[K,N] + bias --------------
// 256 threads = 8 waves in 2x4; workgroup tile 64x128; wave tile 32x32.
// A staged in LDS (one b128 per thread per chunk, reused by 4 waves); B frags
// loaded straight from the frag-major swizzled weights (coalesced 1KB/wave per fragment).
__global__ __launch_bounds__(256) void gemm_bf16(
    const bf16_t* __restrict__ A, const bf16_t* __restrict__ Bsw,
    const float* __restrict__ bias, float* __restrict__ C,
    int N, int K, int lda, int swapBT) {
  __shared__ bf16_t As[64 * 32];
  const int tid = threadIdx.x;
  const int w = tid >> 5, lane = tid & 31;
  const int wm = w >> 2, wn = w & 3;
  const int row0 = blockIdx.x * 64, col0 = blockIdx.y * 128;
  const int n16 = lane & 15, g = lane >> 4;
  const int ntiles = N >> 4;
  const int arow = tid >> 2, acol = (tid & 3) * 8;   // one 16B run per thread

  f32x8 acc[2][2];
#pragma unroll
  for (int jt = 0; jt < 2; ++jt) {
    float bv = bias ? bias[col0 + wn * 32 + jt * 16 + n16] : 0.0f;
#pragma unroll
    for (int it = 0; it < 2; ++it)
#pragma unroll
      for (int r = 0; r < 8; ++r) acc[it][jt][r] = bv;
  }

  for (int k0 = 0; k0 < K; k0 += 32) {
    const int kc = k0 >> 5;
    // stage A 64x32: one contiguous b128 per thread
    *(u32x4*)(As + arow * 32 + acol) =
        *(const u32x4*)(A + (size_t)(row0 + arow) * lda + k0 + acol);
    // prefetch next K-chunk (global_prefetch_b8)
    if (k0 + 32 < K) {
      __builtin_prefetch(A + (size_t)(row0 + arow) * lda + k0 + 32 + acol, 0, 1);
      __builtin_prefetch(Bsw + (((size_t)(kc + 1) * ntiles + (col0 >> 4) + wn * 2) << 9) +
                             lane * 16, 0, 1);
    }
    __syncthreads();
    bf16x16 afr[2], bfr[2];
    afr[0] = load_a_frag(As + (wm * 32) * 32, 32);
    afr[1] = load_a_frag(As + (wm * 32 + 16) * 32, 32);
#pragma unroll
    for (int jt = 0; jt < 2; ++jt)
      bfr[jt] = load_b_frag_swz(Bsw + (((size_t)kc * ntiles + (col0 >> 4) + wn * 2 + jt) << 9));
#pragma unroll
    for (int it = 0; it < 2; ++it)
#pragma unroll
      for (int jt = 0; jt < 2; ++jt)
        acc[it][jt] = wmma_bf16(afr[it], bfr[jt], acc[it][jt]);
    __syncthreads();
  }

#pragma unroll
  for (int it = 0; it < 2; ++it)
#pragma unroll
    for (int jt = 0; jt < 2; ++jt)
#pragma unroll
      for (int r = 0; r < 8; ++r) {
        int row = row0 + wm * 32 + it * 16 + (r + 8 * g);
        int col = col0 + wn * 32 + jt * 16 + n16;
        int orow = swapBT ? ((row & 31) * T_ + (row >> 5)) : row;  // (t*32+b) -> b*T+t
        C[(size_t)orow * N + col] = acc[it][jt][r];
      }
}

// ---- persistent encoder: 1 workgroup, 32 waves, loops S steps --------------
// LDS: z[32][1024] f32 (128KB) + hbf[32][256] bf16 (16KB)
__global__ __launch_bounds__(1024) void encoder_kernel(
    const float* __restrict__ xWx,      // [4096][1024], bias included
    const bf16_t* __restrict__ Whs,     // swizzled [256][1024]
    float* __restrict__ enc_out,        // [4096][256] f32 (row = s*32+b)
    bf16_t* __restrict__ enc_out_bf,    // bf16 copy
    float* __restrict__ c_out) {        // [8192]
  extern __shared__ char smem[];
  float*  zl  = (float*)smem;                          // 32*1024
  bf16_t* hbf = (bf16_t*)(smem + 32 * 1024 * 4);       // 8192
  const int tid = threadIdx.x, w = tid >> 5, lane = tid & 31;
  const int n16 = lane & 15, g = lane >> 4;
  const int mtile = w & 1, ntb = (w >> 1) * 4;

  float c[8];
#pragma unroll
  for (int k = 0; k < 8; ++k) c[k] = 0.0f;
  for (int i = tid; i < 8192; i += 1024) hbf[i] = (bf16_t)0.0f;
  __syncthreads();

  for (int s = 0; s < S_; ++s) {
    f32x8 acc[4];
#pragma unroll
    for (int j = 0; j < 4; ++j) {
      const float* sp = xWx + ((size_t)(s * 32 + mtile * 16)) * G4U + (ntb + j) * 16 + n16;
#pragma unroll
      for (int r = 0; r < 8; ++r) acc[j][r] = sp[(size_t)(r + 8 * g) * G4U];
    }
#pragma unroll 2
    for (int kk = 0; kk < 8; ++kk) {
      bf16x16 a = load_a_frag(hbf + mtile * 16 * U_ + kk * 32, U_);
#pragma unroll
      for (int j = 0; j < 4; ++j) {
        bf16x16 bfr = load_b_frag_swz(Whs + (((size_t)kk * 64 + ntb + j) << 9));
        acc[j] = wmma_bf16(a, bfr, acc[j]);
      }
    }
#pragma unroll
    for (int j = 0; j < 4; ++j)
#pragma unroll
      for (int r = 0; r < 8; ++r)
        zl[(mtile * 16 + r + 8 * g) * G4U + (ntb + j) * 16 + n16] = acc[j][r];
    __syncthreads();

#pragma unroll
    for (int k = 0; k < 8; ++k) {
      int e = tid + k * 1024;
      int b = e >> 8, u = e & 255;
      float zi = zl[b * G4U + u],        zf = zl[b * G4U + 256 + u];
      float zg = zl[b * G4U + 512 + u],  zo = zl[b * G4U + 768 + u];
      float cn = sigmf(zf) * c[k] + sigmf(zi) * tanhf(zg);
      c[k] = cn;
      float h = sigmf(zo) * tanhf(cn);
      hbf[b * U_ + u] = (bf16_t)h;
      size_t o = (size_t)(s * 32 + b) * U_ + u;
      enc_out[o] = h;
      enc_out_bf[o] = (bf16_t)h;
    }
    __syncthreads();
  }
#pragma unroll
  for (int k = 0; k < 8; ++k) c_out[tid + k * 1024] = c[k];
}

// ---- persistent decoder: 1 workgroup, 32 waves, loops T steps --------------
// LDS: zl 128KB (q aliases first 32KB) + sc 16KB + hbf 16KB + ctxbf 16KB
__global__ __launch_bounds__(1024) void decoder_kernel(
    const float* __restrict__ xWx,      // emb-part + bias [4096][1024]
    const bf16_t* __restrict__ WxUs,    // swizzled ctx part [256][1024]
    const bf16_t* __restrict__ Whs,     // swizzled [256][1024]
    const bf16_t* __restrict__ W1s,     // swizzled [256][256]
    const float* __restrict__ b1,       // [256]
    const float* __restrict__ keysT,    // [32][256][128]
    const float* __restrict__ attV,     // [256]
    const float* __restrict__ bV,       // [1]
    const float* __restrict__ enc_out,  // [4096][256] f32
    const float* __restrict__ c_in,     // [8192]
    bf16_t* __restrict__ dec_hs) {      // [4096][256] bf16 (row = t*32+b)
  extern __shared__ char smem[];
  float*  zl    = (float*)smem;                           // 32*1024 (q in [0..8191])
  float*  sc    = (float*)(smem + 131072);                // 32*128
  bf16_t* hbf   = (bf16_t*)(smem + 131072 + 16384);       // 8192
  bf16_t* ctxbf = (bf16_t*)(smem + 131072 + 16384 + 16384);
  const int tid = threadIdx.x, w = tid >> 5, lane = tid & 31;
  const int n16 = lane & 15, g = lane >> 4;
  const float bVv = bV[0];

  float c[8];
#pragma unroll
  for (int k = 0; k < 8; ++k) {
    int e = tid + k * 1024;
    c[k] = c_in[e];
    int b = e >> 8, u = e & 255;
    hbf[e] = (bf16_t)enc_out[(size_t)((S_ - 1) * 32 + b) * U_ + u];  // final enc h
  }
  __syncthreads();

  for (int t = 0; t < T_; ++t) {
    // ---- phase 1: q = h @ W1 + b1  (32 tiles over 32 waves) ----
    {
      const int mt = w & 1, nt = w >> 1;
      f32x8 acc;
      float bb = b1[nt * 16 + n16];
#pragma unroll
      for (int r = 0; r < 8; ++r) acc[r] = bb;
#pragma unroll 2
      for (int kk = 0; kk < 8; ++kk) {
        bf16x16 a = load_a_frag(hbf + mt * 16 * U_ + kk * 32, U_);
        bf16x16 bfr = load_b_frag_swz(W1s + (((size_t)kk * 16 + nt) << 9));
        acc = wmma_bf16(a, bfr, acc);
      }
#pragma unroll
      for (int r = 0; r < 8; ++r)
        zl[(mt * 16 + r + 8 * g) * U_ + nt * 16 + n16] = acc[r];   // q[32][256]
    }
    __syncthreads();
    // ---- phase 2: score[b][s] = sum_u tanh(q+keys)*V + bV ----
#pragma unroll
    for (int i = 0; i < 4; ++i) {
      int p = tid + i * 1024;
      int b = p >> 7, s = p & 127;
      const float* kp = keysT + (size_t)b * U_ * S_ + s;
      const float* qp = zl + b * U_;
      float sum = bVv;
#pragma unroll 4
      for (int u = 0; u < U_; ++u) sum += tanhf(qp[u] + kp[(size_t)u * S_]) * attV[u];
      sc[b * S_ + s] = sum;
    }
    __syncthreads();
    // ---- phase 3: softmax over s (wave w handles b=w) ----
    {
      float v[4], m = -1e30f;
#pragma unroll
      for (int i = 0; i < 4; ++i) { v[i] = sc[w * S_ + lane + i * 32]; m = fmaxf(m, v[i]); }
#pragma unroll
      for (int off = 16; off >= 1; off >>= 1) m = fmaxf(m, __shfl_xor(m, off, 32));
      float ssum = 0.0f;
#pragma unroll
      for (int i = 0; i < 4; ++i) { v[i] = __expf(v[i] - m); ssum += v[i]; }
#pragma unroll
      for (int off = 16; off >= 1; off >>= 1) ssum += __shfl_xor(ssum, off, 32);
      float inv = 1.0f / ssum;
#pragma unroll
      for (int i = 0; i < 4; ++i) sc[w * S_ + lane + i * 32] = v[i] * inv;
    }
    __syncthreads();
    // ---- phase 4: ctx[b][u] = sum_s attn * enc_out ----
#pragma unroll
    for (int k = 0; k < 8; ++k) {
      int e = tid + k * 1024;
      int b = e >> 8, u = e & 255;
      float a = 0.0f;
#pragma unroll 4
      for (int s2 = 0; s2 < S_; ++s2)
        a += sc[b * S_ + s2] * enc_out[(size_t)(s2 * 32 + b) * U_ + u];
      ctxbf[b * U_ + u] = (bf16_t)a;
    }
    __syncthreads();
    // ---- phase 5: z = xWx[t] + ctx@WxU + h@Wh ----
    {
      const int mtile = w & 1, ntb = (w >> 1) * 4;
      f32x8 acc[4];
#pragma unroll
      for (int j = 0; j < 4; ++j) {
        const float* sp = xWx + ((size_t)(t * 32 + mtile * 16)) * G4U + (ntb + j) * 16 + n16;
#pragma unroll
        for (int r = 0; r < 8; ++r) acc[j][r] = sp[(size_t)(r + 8 * g) * G4U];
      }
#pragma unroll 2
      for (int kk = 0; kk < 8; ++kk) {
        bf16x16 a1 = load_a_frag(ctxbf + mtile * 16 * U_ + kk * 32, U_);
        bf16x16 a2 = load_a_frag(hbf + mtile * 16 * U_ + kk * 32, U_);
#pragma unroll
        for (int j = 0; j < 4; ++j) {
          bf16x16 bx = load_b_frag_swz(WxUs + (((size_t)kk * 64 + ntb + j) << 9));
          acc[j] = wmma_bf16(a1, bx, acc[j]);
          bf16x16 bh = load_b_frag_swz(Whs + (((size_t)kk * 64 + ntb + j) << 9));
          acc[j] = wmma_bf16(a2, bh, acc[j]);
        }
      }
#pragma unroll
      for (int j = 0; j < 4; ++j)
#pragma unroll
        for (int r = 0; r < 8; ++r)
          zl[(mtile * 16 + r + 8 * g) * G4U + (ntb + j) * 16 + n16] = acc[j][r];
    }
    __syncthreads();
    // ---- phase 6: LSTM update ----
#pragma unroll
    for (int k = 0; k < 8; ++k) {
      int e = tid + k * 1024;
      int b = e >> 8, u = e & 255;
      float zi = zl[b * G4U + u],       zf = zl[b * G4U + 256 + u];
      float zg = zl[b * G4U + 512 + u], zo = zl[b * G4U + 768 + u];
      float cn = sigmf(zf) * c[k] + sigmf(zi) * tanhf(zg);
      c[k] = cn;
      float h = sigmf(zo) * tanhf(cn);
      hbf[b * U_ + u] = (bf16_t)h;
      dec_hs[(size_t)(t * 32 + b) * U_ + u] = (bf16_t)h;
    }
    __syncthreads();
  }
}

// ---- row softmax over V=32000, one block per (b,t) row ---------------------
__global__ __launch_bounds__(256) void softmax_rows(float* __restrict__ out, int N) {
  __shared__ float red[8];
  float* row = out + (size_t)blockIdx.x * N;
  const int tid = threadIdx.x, w = tid >> 5, lane = tid & 31;
  float m = -1e30f;
  for (int i = tid; i < N; i += 256) m = fmaxf(m, row[i]);
#pragma unroll
  for (int off = 16; off >= 1; off >>= 1) m = fmaxf(m, __shfl_xor(m, off, 32));
  if (lane == 0) red[w] = m;
  __syncthreads();
  float mm = red[0];
#pragma unroll
  for (int j = 1; j < 8; ++j) mm = fmaxf(mm, red[j]);
  __syncthreads();
  float s = 0.0f;
  for (int i = tid; i < N; i += 256) s += __expf(row[i] - mm);
#pragma unroll
  for (int off = 16; off >= 1; off >>= 1) s += __shfl_xor(s, off, 32);
  if (lane == 0) red[w] = s;
  __syncthreads();
  float ss = 0.0f;
#pragma unroll
  for (int j = 0; j < 8; ++j) ss += red[j];
  float inv = 1.0f / ss;
  for (int i = tid; i < N; i += 256) row[i] = __expf(row[i] - mm) * inv;
}

// ---------------------------------------------------------------------------
extern "C" void kernel_launch(void* const* d_in, const int* in_sizes, int n_in,
                              void* d_out, int out_size, void* d_ws, size_t ws_size,
                              hipStream_t stream) {
  const int*   src     = (const int*)d_in[0];
  const int*   tgt     = (const int*)d_in[1];
  const float* enc_emb = (const float*)d_in[2];
  const float* enc_Wx  = (const float*)d_in[3];
  const float* enc_Wh  = (const float*)d_in[4];
  const float* enc_b   = (const float*)d_in[5];
  const float* dec_emb = (const float*)d_in[6];
  const float* dec_Wx  = (const float*)d_in[7];
  const float* dec_Wh  = (const float*)d_in[8];
  const float* dec_b   = (const float*)d_in[9];
  const float* att_W1  = (const float*)d_in[10];
  const float* att_b1  = (const float*)d_in[11];
  const float* att_W2  = (const float*)d_in[12];
  const float* att_b2  = (const float*)d_in[13];
  const float* att_V   = (const float*)d_in[14];
  const float* att_bV  = (const float*)d_in[15];
  const float* fc_W    = (const float*)d_in[16];
  const float* fc_b    = (const float*)d_in[17];
  float* out = (float*)d_out;

  // ---- workspace bump allocator (256B aligned) ----
  char* p = (char*)d_ws;
  auto alloc = [&](size_t bytes) { char* r = p; p += (bytes + 255) & ~(size_t)255; return r; };
  bf16_t* w_encWx  = (bf16_t*)alloc((size_t)E_ * G4U * 2);     // swizzled
  bf16_t* w_encWh  = (bf16_t*)alloc((size_t)U_ * G4U * 2);
  bf16_t* w_decWxU = (bf16_t*)alloc((size_t)U_ * G4U * 2);     // ctx rows 0..255
  bf16_t* w_decWxE = (bf16_t*)alloc((size_t)E_ * G4U * 2);     // emb rows 256..383
  bf16_t* w_decWh  = (bf16_t*)alloc((size_t)U_ * G4U * 2);
  bf16_t* w_attW1  = (bf16_t*)alloc((size_t)U_ * U_ * 2);
  bf16_t* w_attW2  = (bf16_t*)alloc((size_t)U_ * U_ * 2);
  bf16_t* w_fcW    = (bf16_t*)alloc((size_t)U_ * V_ * 2);
  bf16_t* encx     = (bf16_t*)alloc((size_t)4096 * E_ * 2);
  bf16_t* decx     = (bf16_t*)alloc((size_t)4096 * E_ * 2);
  float*  encxWx   = (float*)alloc((size_t)4096 * G4U * 4);
  float*  decxWx   = (float*)alloc((size_t)4096 * G4U * 4);
  float*  encout   = (float*)alloc((size_t)4096 * U_ * 4);
  bf16_t* encoutb  = (bf16_t*)alloc((size_t)4096 * U_ * 2);
  float*  keys     = (float*)alloc((size_t)4096 * U_ * 4);
  float*  keysT    = (float*)alloc((size_t)4096 * U_ * 4);
  bf16_t* dechs    = (bf16_t*)alloc((size_t)4096 * U_ * 2);
  float*  cbuf     = (float*)alloc((size_t)8192 * 4);

  auto swz = [&](const float* s0, bf16_t* d0, int K, int N) {
    int pairs = (K * N) >> 1;
    cvt_swz<<<(pairs + 255) / 256, 256, 0, stream>>>(s0, d0, K, N);
  };
  swz(enc_Wx, w_encWx, E_, G4U);
  swz(enc_Wh, w_encWh, U_, G4U);
  swz(dec_Wx, w_decWxU, U_, G4U);                     // rows 0..255 (ctx)
  swz(dec_Wx + (size_t)U_ * G4U, w_decWxE, E_, G4U);  // rows 256..383 (emb)
  swz(dec_Wh, w_decWh, U_, G4U);
  swz(att_W1, w_attW1, U_, U_);
  swz(att_W2, w_attW2, U_, U_);
  swz(fc_W,   w_fcW,   U_, V_);

  gather_enc<<<4096, 128, 0, stream>>>(src, enc_emb, encx);
  gather_dec<<<4096, 128, 0, stream>>>(tgt, dec_emb, decx);

  // enc_xWx = encx @ enc_Wx + enc_b      [4096,128]x[128,1024]
  gemm_bf16<<<dim3(64, 8), 256, 0, stream>>>(encx, w_encWx, enc_b, encxWx,
                                             G4U, E_, E_, 0);
  // dec_xWx = decx @ dec_Wx[U:] + dec_b  [4096,128]x[128,1024]
  gemm_bf16<<<dim3(64, 8), 256, 0, stream>>>(decx, w_decWxE, dec_b, decxWx,
                                             G4U, E_, E_, 0);
  // encoder recurrence (1 WGP, 32 waves, 144KB LDS)
  encoder_kernel<<<1, 1024, 147456, stream>>>(encxWx, w_encWh, encout, encoutb, cbuf);
  // keys = enc_out @ att_W2 + att_b2     [4096,256]x[256,256]
  gemm_bf16<<<dim3(64, 2), 256, 0, stream>>>(encoutb, w_attW2, att_b2, keys,
                                             U_, U_, U_, 0);
  transpose_keys<<<4096, 256, 0, stream>>>(keys, keysT);
  // decoder recurrence (1 WGP, 176KB LDS)
  decoder_kernel<<<1, 1024, 180224, stream>>>(decxWx, w_decWxU, w_decWh, w_attW1,
                                              att_b1, keysT, att_V, att_bV,
                                              encout, cbuf, dechs);
  // logits = dec_hs @ fc_W + fc_b -> d_out with [B,T,V] layout (swapBT)
  gemm_bf16<<<dim3(64, 250), 256, 0, stream>>>(dechs, w_fcW, fc_b, out,
                                               V_, U_, U_, 1);
  softmax_rows<<<4096, 256, 0, stream>>>(out, V_);
}
